// MultiHeadAttentionWithRoPE_30339648979645
// MI455X (gfx1250) — matrix operand order repaired and checked by reading, not compiled, not run on hardware
//
#include <hip/hip_runtime.h>
#include <hip/hip_bf16.h>
#include <cstdint>

typedef __bf16 bf16_t;
typedef __attribute__((ext_vector_type(16))) __bf16 v16bf;
typedef __attribute__((ext_vector_type(8)))  __bf16 v8bf;
typedef __attribute__((ext_vector_type(4)))  __bf16 v4bf;
typedef __attribute__((ext_vector_type(8)))  float  v8f;
typedef __attribute__((ext_vector_type(4)))  int    v4i;

#define D_MODEL   2048
#define NUM_HEADS 16
#define HEAD_DIM  128
#define BATCH     2
#define SEQ       2048
#define MTOT      (BATCH * SEQ)

// ---------------------------------------------------------------------------
// CDNA5 async global->LDS copy (ASYNCcnt-tracked), guarded with a synchronous
// fallback. Builtin signature (verified by round-2 diagnostic): int4 pointers.
// ---------------------------------------------------------------------------
#if __has_builtin(__builtin_amdgcn_global_load_async_to_lds_b128)
#define HAS_ASYNC_LDS 1
#else
#define HAS_ASYNC_LDS 0
#endif

typedef __attribute__((address_space(1))) v4i* as1_v4i_ptr;
typedef __attribute__((address_space(3))) v4i* as3_v4i_ptr;

__device__ __forceinline__ void cp16_g2l(const bf16_t* g, bf16_t* l) {
#if HAS_ASYNC_LDS
  as1_v4i_ptr gp = reinterpret_cast<as1_v4i_ptr>(reinterpret_cast<uintptr_t>(g));
  as3_v4i_ptr lp = reinterpret_cast<as3_v4i_ptr>(reinterpret_cast<uintptr_t>(l));
  __builtin_amdgcn_global_load_async_to_lds_b128(gp, lp, 0, 0);
#else
  *(v8bf*)l = *(const v8bf*)g;
#endif
}

__device__ __forceinline__ void wait_async0() {
#if HAS_ASYNC_LDS && __has_builtin(__builtin_amdgcn_s_wait_asynccnt)
  __builtin_amdgcn_s_wait_asynccnt(0);
#endif
}

// ---------------------------------------------------------------------------
// GEMM: C[M x N] = X[M x K] @ W[N x K]^T + bias[N]   (torch Linear semantics)
// Tile: BM=64 x BN=128 x BK=32. 128 threads = 4 waves; wave w owns rows
// [16w,16w+16) x all 128 columns -> 8 WMMAs per K-step per wave.
// MODE 0: bf16 out, [B,H,S,hd] scatter (Q,K)
// MODE 1: bf16 out, [B,H,hd,S] scatter (V transposed for the PV GEMM)
// MODE 2: f32  out, row-major [M,N]    (final projection)
// ---------------------------------------------------------------------------
template <typename TIN, int MODE>
__global__ __launch_bounds__(128) void gemm_wmma(const TIN* __restrict__ X,
                                                 const float* __restrict__ W,
                                                 const float* __restrict__ bias,
                                                 void* __restrict__ outv) {
  constexpr int BM = 64, BN = 128, BK = 32, LDK = 40;  // 80B rows, 16B aligned
  __shared__ __align__(16) bf16_t Xs[BM * LDK];
  __shared__ __align__(16) bf16_t Ws[BN * LDK];

  const int tid   = threadIdx.x;
  const int wave  = tid >> 5;
  const int lane  = tid & 31;
  const int lhalf = lane >> 4;
  const int lr    = lane & 15;
  const int m0 = blockIdx.x * BM;
  const int n0 = blockIdx.y * BN;
  const int mw = wave * 16;

  v8f acc[8] = {};

  for (int k0 = 0; k0 < D_MODEL; k0 += BK) {
    // Stage X tile (64x32) as bf16.
#pragma unroll
    for (int i = 0; i < 4; ++i) {
      const int g   = tid + i * 128;   // 0..511
      const int row = g >> 3;
      const int c4  = (g & 7) * 4;
      v4bf xv;
      if constexpr (sizeof(TIN) == 4) {
        const float4 f = *(const float4*)(X + (size_t)(m0 + row) * D_MODEL + k0 + c4);
        xv[0] = (bf16_t)f.x; xv[1] = (bf16_t)f.y; xv[2] = (bf16_t)f.z; xv[3] = (bf16_t)f.w;
      } else {
        xv = *(const v4bf*)(X + (size_t)(m0 + row) * D_MODEL + k0 + c4);
      }
      *(v4bf*)&Xs[row * LDK + c4] = xv;
      if (k0 + BK < D_MODEL)
        __builtin_prefetch(X + (size_t)(m0 + row) * D_MODEL + k0 + BK + c4, 0, 0);
    }
    // Stage W tile (128x32) as bf16.
#pragma unroll
    for (int i = 0; i < 8; ++i) {
      const int g   = tid + i * 128;   // 0..1023
      const int row = g >> 3;
      const int c4  = (g & 7) * 4;
      const float4 wf = *(const float4*)(W + (size_t)(n0 + row) * D_MODEL + k0 + c4);
      v4bf wv;
      wv[0] = (bf16_t)wf.x; wv[1] = (bf16_t)wf.y; wv[2] = (bf16_t)wf.z; wv[3] = (bf16_t)wf.w;
      *(v4bf*)&Ws[row * LDK + c4] = wv;
      if (k0 + BK < D_MODEL)
        __builtin_prefetch(W + (size_t)(n0 + row) * D_MODEL + k0 + BK + c4, 0, 0);
    }
    __syncthreads();

    // A fragment 16x32: lanes 0-15 -> K 0-7/16-23, lanes 16-31 -> K 8-15/24-31
    v16bf a;
    {
      const v8bf lo = *(const v8bf*)&Xs[(mw + lr) * LDK + lhalf * 8];
      const v8bf hi = *(const v8bf*)&Xs[(mw + lr) * LDK + 16 + lhalf * 8];
#pragma unroll
      for (int i = 0; i < 8; ++i) { a[i] = lo[i]; a[8 + i] = hi[i]; }
    }
    // Load B fragments in groups of 4, then issue 4 WMMAs, so DS loads can
    // clause/overlap instead of draining before every matrix op.
#pragma unroll
    for (int g4 = 0; g4 < 2; ++g4) {
      v16bf b[4];
#pragma unroll
      for (int q = 0; q < 4; ++q) {
        const int nt = g4 * 4 + q;
        const v8bf lo = *(const v8bf*)&Ws[(nt * 16 + lr) * LDK + lhalf * 8];
        const v8bf hi = *(const v8bf*)&Ws[(nt * 16 + lr) * LDK + 16 + lhalf * 8];
#pragma unroll
        for (int i = 0; i < 8; ++i) { b[q][i] = lo[i]; b[q][8 + i] = hi[i]; }
      }
#pragma unroll
      for (int q = 0; q < 4; ++q) {
        const int nt = g4 * 4 + q;
        acc[nt] = __builtin_amdgcn_wmma_f32_16x16x32_bf16(false, a, false, b[q], (short)0,
                                                          acc[nt], false, false);
      }
    }
    __syncthreads();
  }

  // Epilogue. C layout: VGPR r <-> row (lhalf*8 + r), column <-> (lane&15).
#pragma unroll
  for (int nt = 0; nt < 8; ++nt) {
#pragma unroll
    for (int r = 0; r < 8; ++r) {
      const int mrow = m0 + mw + lhalf * 8 + r;
      const int ncol = n0 + nt * 16 + lr;
      const float v  = acc[nt][r] + bias[ncol];
      const int bb = mrow / SEQ, s = mrow % SEQ;
      const int h = ncol / HEAD_DIM, d = ncol % HEAD_DIM;
      if constexpr (MODE == 0) {
        ((bf16_t*)outv)[(((size_t)bb * NUM_HEADS + h) * SEQ + s) * HEAD_DIM + d] = (bf16_t)v;
      } else if constexpr (MODE == 1) {
        ((bf16_t*)outv)[(((size_t)bb * NUM_HEADS + h) * HEAD_DIM + d) * SEQ + s] = (bf16_t)v;
      } else {
        ((float*)outv)[(size_t)mrow * D_MODEL + ncol] = v;
      }
    }
  }
}

// ---------------------------------------------------------------------------
// RoPE applied in-place on bf16 Q,K stored [B,H,S,hd]; pairs (j, j+64).
// ---------------------------------------------------------------------------
__global__ __launch_bounds__(256) void rope_kernel(bf16_t* __restrict__ Q,
                                                   bf16_t* __restrict__ K,
                                                   const int* __restrict__ pos_ids) {
  const int tid = blockIdx.x * 256 + threadIdx.x;     // B*H*S*64 threads
  const int j  = tid & 63;
  const int s  = (tid >> 6) & (SEQ - 1);
  const int bh = tid >> 17;
  if (bh >= BATCH * NUM_HEADS) return;
  const int b   = bh >> 4;  // / NUM_HEADS
  const int pos = pos_ids[b * SEQ + s];
  const float ang = (float)pos * __expf((float)(-2 * j) * (9.210340371976184f / (float)HEAD_DIM));
  const float c = __cosf(ang), sn = __sinf(ang);
  const size_t base = ((size_t)bh * SEQ + s) * HEAD_DIM;
  const float q1 = (float)Q[base + j], q2 = (float)Q[base + j + 64];
  Q[base + j]      = (bf16_t)(q1 * c - q2 * sn);
  Q[base + j + 64] = (bf16_t)(q1 * sn + q2 * c);
  const float k1 = (float)K[base + j], k2 = (float)K[base + j + 64];
  K[base + j]      = (bf16_t)(k1 * c - k2 * sn);
  K[base + j + 64] = (bf16_t)(k1 * sn + k2 * c);
}

// ---------------------------------------------------------------------------
// Flash attention. Grid (S/64, B*H); 128 threads = 4 waves; each wave owns 16
// query rows. K (32x128) and V ([hd,S]-transposed, 128x32) tiles are staged
// into double-buffered LDS with async global->LDS copies; tile t+1 transfer
// overlaps tile t compute. The softmax denominator is computed on the matrix
// pipe: rowsum(P) = P @ ones via one extra WMMA per tile. P goes C-layout ->
// A-layout via a per-wave private LDS tile (same-wave DS ops are in-order).
// All fragment loops are force-unrolled so every accumulator array is
// statically indexed (avoids v_movrel traffic).
// ---------------------------------------------------------------------------
__global__ __launch_bounds__(128) void flash_attn(const bf16_t* __restrict__ Q,
                                                  const bf16_t* __restrict__ K,
                                                  const bf16_t* __restrict__ Vt,
                                                  bf16_t* __restrict__ ctx) {
  constexpr int KLD = 136;  // 32x128 K tile rows, 272B stride (16B aligned)
  constexpr int VLD = 40;   // 128x32 V tile rows, 80B stride  (16B aligned)
  constexpr int PLD = 40;
  __shared__ __align__(16) bf16_t Ks[2][32 * KLD];
  __shared__ __align__(16) bf16_t Vs[2][128 * VLD];
  __shared__ __align__(16) bf16_t Pls[4 * 16 * PLD];

  const int tid = threadIdx.x;
  const int wave = tid >> 5, lane = tid & 31, lhalf = lane >> 4, lr = lane & 15;
  const int bh = blockIdx.y;
  const int m0 = blockIdx.x * 64 + wave * 16;
  const bf16_t* Qb = Q  + (size_t)bh * SEQ * HEAD_DIM;
  const bf16_t* Kb = K  + (size_t)bh * SEQ * HEAD_DIM;
  const bf16_t* Vb = Vt + (size_t)bh * HEAD_DIM * SEQ;
  bf16_t* Pw = &Pls[wave * 16 * PLD];

  // Stage one 32-key tile: K 32x128 (4 chunks/thread), V 128x32 (4 chunks/thread).
  auto stage_tile = [&](int k0, int buf) {
#pragma unroll
    for (int i = 0; i < 4; ++i) {
      const int c    = tid + i * 128;        // 0..511
      const int krow = c >> 4;               // K: 16 x 8-elt chunks per row
      const int kco  = (c & 15) * 8;
      cp16_g2l(Kb + (size_t)(k0 + krow) * HEAD_DIM + kco, &Ks[buf][krow * KLD + kco]);
      const int vrow = c >> 2;               // V: 4 x 8-elt chunks per row
      const int vco  = (c & 3) * 8;
      cp16_g2l(Vb + (size_t)vrow * SEQ + k0 + vco, &Vs[buf][vrow * VLD + vco]);
    }
  };

  // Q A-fragments: 4 chunks across hd=128 (register resident for whole kernel).
  v16bf qf[4];
#pragma unroll
  for (int c = 0; c < 4; ++c) {
    const v8bf lo = *(const v8bf*)(Qb + (size_t)(m0 + lr) * HEAD_DIM + c * 32 + lhalf * 8);
    const v8bf hi = *(const v8bf*)(Qb + (size_t)(m0 + lr) * HEAD_DIM + c * 32 + 16 + lhalf * 8);
#pragma unroll
    for (int i = 0; i < 8; ++i) { qf[c][i] = lo[i]; qf[c][8 + i] = hi[i]; }
  }

  v16bf ones;               // all-ones 32x16 B fragment for rowsum(P) = P @ 1
#pragma unroll
  for (int i = 0; i < 16; ++i) ones[i] = (bf16_t)1.0f;

  float rm[8];
  v8f oacc[8] = {};
  v8f osum = {};            // running softmax denominator (every column equal)
#pragma unroll
  for (int r = 0; r < 8; ++r) rm[r] = -3.0e38f;
  const float scl = 0.08838834764831845f;  // 1/sqrt(128)

  stage_tile(0, 0);

  for (int kt = 0; kt < SEQ / 32; ++kt) {
    const int buf = kt & 1;
    wait_async0();      // our async copies into buf have landed
    __syncthreads();    // ...and everyone is done reading buf^1 from last round
    if (kt + 1 < SEQ / 32) stage_tile((kt + 1) * 32, buf ^ 1);

    // scores tile: 16 queries x 32 keys, contraction over hd (4 WMMA steps).
    // Fragments loaded 4-at-a-time into distinct regs so DS loads overlap WMMA.
    v8f sf[2] = {};
#pragma unroll
    for (int ng = 0; ng < 2; ++ng) {
      const bf16_t* krow = &Ks[buf][(ng * 16 + lr) * KLD];
      v16bf bfr[4];
#pragma unroll
      for (int c = 0; c < 4; ++c) {
        const v8bf lo = *(const v8bf*)(krow + c * 32 + lhalf * 8);
        const v8bf hi = *(const v8bf*)(krow + c * 32 + 16 + lhalf * 8);
#pragma unroll
        for (int i = 0; i < 8; ++i) { bfr[c][i] = lo[i]; bfr[c][8 + i] = hi[i]; }
      }
#pragma unroll
      for (int c = 0; c < 4; ++c)
        sf[ng] = __builtin_amdgcn_wmma_f32_16x16x32_bf16(false, qf[c], false, bfr[c], (short)0,
                                                         sf[ng], false, false);
    }
    // Online softmax max: C-layout row r lives in VGPR r across one 16-lane
    // half, so a 16-wide xor-shuffle tree reduces exactly one row.
    float fac[8];
#pragma unroll
    for (int r = 0; r < 8; ++r) {
      float v = fmaxf(sf[0][r], sf[1][r]) * scl;
#pragma unroll
      for (int off = 8; off >= 1; off >>= 1) v = fmaxf(v, __shfl_xor(v, off, 32));
      const float nm = fmaxf(rm[r], v);
      fac[r] = __expf(rm[r] - nm);
      rm[r]  = nm;
    }
#pragma unroll
    for (int ng = 0; ng < 2; ++ng) {
#pragma unroll
      for (int r = 0; r < 8; ++r) {
        const float p = __expf(sf[ng][r] * scl - rm[r]);
        Pw[(lhalf * 8 + r) * PLD + ng * 16 + lr] = (bf16_t)p;  // C-layout -> LDS
      }
    }
    // Reload P in A-fragment layout (wave-private LDS; same-wave DS is in-order).
    v16bf pf;
    {
      const v8bf lo = *(const v8bf*)&Pw[lr * PLD + lhalf * 8];
      const v8bf hi = *(const v8bf*)&Pw[lr * PLD + 16 + lhalf * 8];
#pragma unroll
      for (int i = 0; i < 8; ++i) { pf[i] = lo[i]; pf[8 + i] = hi[i]; }
    }
    // Denominator on the matrix pipe: osum = osum*fac + P @ ones.
    {
      v8f o = osum;
#pragma unroll
      for (int r = 0; r < 8; ++r) o[r] *= fac[r];
      osum = __builtin_amdgcn_wmma_f32_16x16x32_bf16(false, pf, false, ones, (short)0,
                                                     o, false, false);
    }
    // O = O*fac + P @ V ; V B-fragments contiguous thanks to [hd,S] layout.
#pragma unroll
    for (int half = 0; half < 2; ++half) {
      v16bf vf[4];
#pragma unroll
      for (int q = 0; q < 4; ++q) {
        const int nt = half * 4 + q;
        const bf16_t* vrow = &Vs[buf][(nt * 16 + lr) * VLD + lhalf * 8];
        const v8bf lo = *(const v8bf*)(vrow);
        const v8bf hi = *(const v8bf*)(vrow + 16);
#pragma unroll
        for (int i = 0; i < 8; ++i) { vf[q][i] = lo[i]; vf[q][8 + i] = hi[i]; }
      }
#pragma unroll
      for (int q = 0; q < 4; ++q) {
        const int nt = half * 4 + q;
        v8f o = oacc[nt];
#pragma unroll
        for (int r = 0; r < 8; ++r) o[r] *= fac[r];
        oacc[nt] = __builtin_amdgcn_wmma_f32_16x16x32_bf16(false, pf, false, vf[q], (short)0,
                                                           o, false, false);
      }
    }
  }

  // Normalize and scatter to ctx [B,S,D].
  const int b = bh / NUM_HEADS, h = bh % NUM_HEADS;
#pragma unroll
  for (int nt = 0; nt < 8; ++nt) {
#pragma unroll
    for (int r = 0; r < 8; ++r) {
      const float v = oacc[nt][r] / osum[r];
      const int s = m0 + lhalf * 8 + r;
      const int d = nt * 16 + lr;
      ctx[((size_t)b * SEQ + s) * D_MODEL + h * HEAD_DIM + d] = (bf16_t)v;
    }
  }
}

// ---------------------------------------------------------------------------
extern "C" void kernel_launch(void* const* d_in, const int* in_sizes, int n_in,
                              void* d_out, int out_size, void* d_ws, size_t ws_size,
                              hipStream_t stream) {
  const float* x  = (const float*)d_in[0];
  const int* pos  = (const int*)d_in[1];
  const float* Wq = (const float*)d_in[2];
  const float* bq = (const float*)d_in[3];
  const float* Wk = (const float*)d_in[4];
  const float* bk = (const float*)d_in[5];
  const float* Wv = (const float*)d_in[6];
  const float* bv = (const float*)d_in[7];
  const float* Wo = (const float*)d_in[8];
  const float* bo = (const float*)d_in[9];
  float* out = (float*)d_out;

  const size_t QN = (size_t)BATCH * NUM_HEADS * SEQ * HEAD_DIM;  // elements
  bf16_t* Qr  = (bf16_t*)d_ws;
  bf16_t* Kr  = Qr + QN;
  bf16_t* Vt  = Kr + QN;
  bf16_t* ctx = Vt + QN;

  const dim3 gg(MTOT / 64, D_MODEL / 128);
  gemm_wmma<float, 0><<<gg, 128, 0, stream>>>(x, Wq, bq, Qr);
  gemm_wmma<float, 0><<<gg, 128, 0, stream>>>(x, Wk, bk, Kr);
  gemm_wmma<float, 1><<<gg, 128, 0, stream>>>(x, Wv, bv, Vt);

  const int nrope = BATCH * NUM_HEADS * SEQ * 64;
  rope_kernel<<<nrope / 256, 256, 0, stream>>>(Qr, Kr, pos);

  flash_attn<<<dim3(SEQ / 64, BATCH * NUM_HEADS), 128, 0, stream>>>(Qr, Kr, Vt, ctx);

  gemm_wmma<bf16_t, 2><<<gg, 128, 0, stream>>>(ctx, Wo, bo, out);
}
